// CausalSelfAttention_28707561406800
// MI455X (gfx1250) — matrix operand order repaired and checked by reading, not compiled, not run on hardware
//
#include <hip/hip_runtime.h>
#include <cstdint>
#include <cstddef>

// ---------------------------------------------------------------------------
// CDNA5 (gfx1250) causal self-attention: bf16 WMMA GEMMs + fp32 flash softmax
// Async global->LDS double-buffered GEMM staging (ASYNCcnt), TN data layout.
// ---------------------------------------------------------------------------

typedef __attribute__((ext_vector_type(16))) __bf16 v16bf;
typedef __attribute__((ext_vector_type(8)))  float  v8f;

constexpr int kB  = 2;
constexpr int kT  = 2048;
constexpr int kC  = 1024;
constexpr int kNH = 16;
constexpr int kHD = 64;   // kC / kNH

// ---- gfx1250 async global->LDS copy (builtin confirmed present) ------------
#if defined(__AMDGCN__) && __has_builtin(__builtin_amdgcn_global_load_async_to_lds_b128)
#define HAVE_ASYNC_LDS 1
#else
#define HAVE_ASYNC_LDS 0
#endif

#if HAVE_ASYNC_LDS
typedef int v4i __attribute__((vector_size(16)));
typedef __attribute__((address_space(1))) v4i* as1_v4i_ptr;
typedef __attribute__((address_space(3))) v4i* as3_v4i_ptr;
// Global pointers: generic and AS1 are bit-identical -> inttoptr.
// LDS pointers: AS3 address is the low 32 bits of the generic address.
#define CP_B128(gp, lp)                                                        \
  __builtin_amdgcn_global_load_async_to_lds_b128(                              \
      (as1_v4i_ptr)(uintptr_t)(gp),                                            \
      (as3_v4i_ptr)(uint32_t)(uintptr_t)(lp), 0, 0)
#if __has_builtin(__builtin_amdgcn_s_wait_asynccnt)
#define ASYNC_WAIT() __builtin_amdgcn_s_wait_asynccnt(0)
#else
#define ASYNC_WAIT() asm volatile("s_wait_asynccnt 0x0" ::: "memory")
#endif
#else
#define CP_B128(gp, lp) (*(uint4*)(lp) = *(const uint4*)(gp))
#define ASYNC_WAIT() ((void)0)
#endif

__device__ inline v8f wmma_bf16(v16bf a, v16bf b, v8f c) {
  // D(16x16,f32) = A(16x32,bf16) * B(32x16,bf16) + C
  return __builtin_amdgcn_wmma_f32_16x16x32_bf16(
      false, a, false, b, (short)0, c, false, false);
}

__device__ inline v8f zero8() {
  v8f z = {0.f, 0.f, 0.f, 0.f, 0.f, 0.f, 0.f, 0.f};
  return z;
}

// A fragment (16x32 bf16) from row-major [16][ld] memory (LDS or global).
// Contiguous k order (matches B loader): lane L -> row M=L%16; halves h -> K = 16*(L>>4)+h.
// Two 16B-aligned b128 loads per lane.
__device__ inline v16bf load_A_rowmajor(const __bf16* mem, int ld, int lane) {
  int m  = lane & 15;
  int qi = lane >> 4;
  union { v16bf v; uint4 q[2]; } r;
  const uint4* p4 = (const uint4*)(mem + (size_t)m * ld);
  r.q[0] = p4[2 * qi];
  r.q[1] = p4[2 * qi + 1];
  return r.v;
}

// B fragment (32x16 bf16) from N-major memory: element (K,N) at mem[N*ld+K].
// Contiguous k order (as A): lane L -> col N=L%16; halves h -> K = 16*(L>>4)+h. Two b128 per lane.
__device__ inline v16bf load_B_nmajor(const __bf16* mem, int ld, int lane) {
  int n  = lane & 15;
  int kb = (lane >> 4) << 4;
  union { v16bf v; uint4 q[2]; } r;
  const uint4* p4 = (const uint4*)(mem + (size_t)n * ld + kb);
  r.q[0] = p4[0];
  r.q[1] = p4[1];
  return r.v;
}

__device__ inline float half16_reduce_max(float x) {
  x = fmaxf(x, __shfl_xor(x, 1, 32));
  x = fmaxf(x, __shfl_xor(x, 2, 32));
  x = fmaxf(x, __shfl_xor(x, 4, 32));
  x = fmaxf(x, __shfl_xor(x, 8, 32));
  return x;
}
__device__ inline float half16_reduce_sum(float x) {
  x += __shfl_xor(x, 1, 32);
  x += __shfl_xor(x, 2, 32);
  x += __shfl_xor(x, 4, 32);
  x += __shfl_xor(x, 8, 32);
  return x;
}

// ---------------------------------------------------------------------------
// fp32 -> bf16 elementwise cast
// ---------------------------------------------------------------------------
__global__ void cast_f32_to_bf16(const float* __restrict__ in,
                                 __bf16* __restrict__ out, int n) {
  int i = blockIdx.x * blockDim.x + threadIdx.x;
  if (i < n) out[i] = (__bf16)in[i];
}

// ---------------------------------------------------------------------------
// fp32 [K][N] -> bf16 transposed [N][K] (one-time weight re-layout so GEMM
// B-fragments are K-contiguous per lane; no per-stage transpose in hot loop).
// Block 256 = 32x8, 32x32 tile via padded LDS.
// ---------------------------------------------------------------------------
__global__ __launch_bounds__(256)
void cast_transpose_f32_bf16(const float* __restrict__ in,
                             __bf16* __restrict__ out, int K, int N) {
  __shared__ float tile[32][33];
  const int k0 = blockIdx.y * 32, n0 = blockIdx.x * 32;
  const int tx = threadIdx.x & 31;
  const int ty = threadIdx.x >> 5;   // 0..7
#pragma unroll
  for (int i = 0; i < 4; ++i) {
    int k = ty + i * 8;
    tile[k][tx] = in[(size_t)(k0 + k) * N + n0 + tx];
  }
  __syncthreads();
#pragma unroll
  for (int i = 0; i < 4; ++i) {
    int n = ty + i * 8;
    out[(size_t)(n0 + n) * K + k0 + tx] = (__bf16)tile[tx][n];
  }
}

// ---------------------------------------------------------------------------
// bf16 GEMM, TN layout: C(MxN,f32) = A(MxK, row-major) * Bt(NxK, N-major)^T.
// Block 256 threads = 8 waves as 4(M) x 2(N); block tile 128x128, K-stage 64.
// Double-buffered LDS, tiles staged with async global->LDS b128.
// Per wave per stage: 16 WMMA vs 24 b128 LDS fragment loads.
// ---------------------------------------------------------------------------
__global__ __launch_bounds__(256)
void gemm_bf16_tn(const __bf16* __restrict__ A, const __bf16* __restrict__ Bt,
                  float* __restrict__ Cm, int M, int N, int K) {
  __shared__ __align__(16) __bf16 at[2][128 * 64];   // [row m][k]
  __shared__ __align__(16) __bf16 bt[2][128 * 64];   // [row n][k]

  const int tid  = threadIdx.x;
  const int wave = tid >> 5;
  const int lane = tid & 31;
  const int m0   = blockIdx.y * 128;
  const int n0   = blockIdx.x * 128;
  const int wm   = (wave & 3) * 32;   // wave's M offset inside tile
  const int wn   = (wave >> 2) * 64;  // wave's N offset inside tile

  // staging coords: 2 threads per row, 32 halfs (4 x b128) each
  const int srow = tid >> 1;
  const int scol = (tid & 1) * 32;

  v8f acc[2][4];
#pragma unroll
  for (int mi = 0; mi < 2; ++mi)
#pragma unroll
    for (int ni = 0; ni < 4; ++ni) acc[mi][ni] = zero8();

  const int nk = K >> 6;  // 64-deep K stages

  // prologue: stage k-tile 0 into buffer 0
  {
    const __bf16* ga = A  + (size_t)(m0 + srow) * K + scol;
    const __bf16* gb = Bt + (size_t)(n0 + srow) * K + scol;
    __bf16* la = at[0] + srow * 64 + scol;
    __bf16* lb = bt[0] + srow * 64 + scol;
#pragma unroll
    for (int i = 0; i < 4; ++i) {
      CP_B128(ga + i * 8, la + i * 8);
      CP_B128(gb + i * 8, lb + i * 8);
    }
  }

  for (int kk = 0; kk < nk; ++kk) {
    ASYNC_WAIT();
    __syncthreads();

    const int cur = kk & 1;
    if (kk + 1 < nk) {
      const int nxt = cur ^ 1;
      const int ko  = (kk + 1) << 6;
      const __bf16* ga = A  + (size_t)(m0 + srow) * K + ko + scol;
      const __bf16* gb = Bt + (size_t)(n0 + srow) * K + ko + scol;
      __bf16* la = at[nxt] + srow * 64 + scol;
      __bf16* lb = bt[nxt] + srow * 64 + scol;
#pragma unroll
      for (int i = 0; i < 4; ++i) {
        CP_B128(ga + i * 8, la + i * 8);
        CP_B128(gb + i * 8, lb + i * 8);
      }
    }

    const __bf16* ab = at[cur] + wm * 64;
    const __bf16* bb = bt[cur] + wn * 64;
#pragma unroll
    for (int kc = 0; kc < 2; ++kc) {
      v16bf a0 = load_A_rowmajor(ab + 0 * 64 + kc * 32, 64, lane);
      v16bf a1 = load_A_rowmajor(ab + 16 * 64 + kc * 32, 64, lane);
#pragma unroll
      for (int ni = 0; ni < 4; ++ni) {
        v16bf b = load_B_nmajor(bb + ni * 16 * 64 + kc * 32, 64, lane);
        acc[0][ni] = wmma_bf16(a0, b, acc[0][ni]);
        acc[1][ni] = wmma_bf16(a1, b, acc[1][ni]);
      }
    }
    __syncthreads();
  }

  // Epilogue: one row base per output row, constant column offsets.
  const int n  = lane & 15;
  const int mb = (lane >> 4) << 3;
#pragma unroll
  for (int mi = 0; mi < 2; ++mi)
#pragma unroll
    for (int v = 0; v < 8; ++v) {
      float* rp = Cm + (size_t)(m0 + wm + mi * 16 + mb + v) * N + n0 + wn + n;
#pragma unroll
      for (int ni = 0; ni < 4; ++ni) rp[ni * 16] = acc[mi][ni][v];
    }
}

// ---------------------------------------------------------------------------
// RoPE + head split.
//  qkv fp32 [B,T,3C] -> q,k bf16 [B,NH,T,HD] (q pre-scaled by 1/sqrt(HD)),
//                       v  bf16 [B,NH,HD,T] (transposed for B-fragments).
// ---------------------------------------------------------------------------
__global__ void rope_split_kernel(const float* __restrict__ qkv,
                                  const float* __restrict__ cosb,
                                  const float* __restrict__ sinb,
                                  __bf16* __restrict__ qo,
                                  __bf16* __restrict__ ko,
                                  __bf16* __restrict__ vo) {
  int idx = blockIdx.x * blockDim.x + threadIdx.x;  // < B*T*NH*(HD/2)
  int d2 = idx & 31;
  int h  = (idx >> 5) & (kNH - 1);
  int t  = (idx >> 9) & (kT - 1);
  int b  = idx >> 20;

  const float* row = qkv + ((size_t)b * kT + t) * (3 * kC);
  int d0 = 2 * d2, d1 = 2 * d2 + 1;
  float q0 = row[h * kHD + d0],          q1 = row[h * kHD + d1];
  float k0 = row[kC + h * kHD + d0],     k1 = row[kC + h * kHD + d1];
  float v0 = row[2 * kC + h * kHD + d0], v1 = row[2 * kC + h * kHD + d1];

  float c = cosb[t * kHD + d0];
  float s = sinb[t * kHD + d0];

  float qr0 = q0 * c - q1 * s, qr1 = q1 * c + q0 * s;
  float kr0 = k0 * c - k1 * s, kr1 = k1 * c + k0 * s;
  const float scale = 0.125f;  // 1/sqrt(64)

  size_t bh   = (size_t)b * kNH + h;
  size_t base = (bh * kT + t) * kHD;
  qo[base + d0] = (__bf16)(qr0 * scale);
  qo[base + d1] = (__bf16)(qr1 * scale);
  ko[base + d0] = (__bf16)kr0;
  ko[base + d1] = (__bf16)kr1;
  vo[(bh * kHD + d0) * kT + t] = (__bf16)v0;
  vo[(bh * kHD + d1) * kT + t] = (__bf16)v1;
}

// ---------------------------------------------------------------------------
// Causal flash attention. Grid (T/128, NH, B), 256 threads = 8 waves.
// Wave owns 16 query rows; streams 32-key tiles with online softmax.
// Per tile: 4 WMMA (S=Q K^T) + 4 WMMA (O+=P V); fp32 row stats via shfl.
// ---------------------------------------------------------------------------
__global__ __launch_bounds__(256)
void flash_attn_kernel(const __bf16* __restrict__ qg,
                       const __bf16* __restrict__ kg,
                       const __bf16* __restrict__ vtg,
                       __bf16* __restrict__ yg) {
  __shared__ __align__(16) __bf16 lds_p[8][16 * 32];  // per-wave P scratch

  const int wave = threadIdx.x >> 5;
  const int lane = threadIdx.x & 31;
  const int h = blockIdx.y;
  const int b = blockIdx.z;
  const size_t bh = (size_t)b * kNH + h;

  const __bf16* q  = qg  + bh * kT * kHD;   // [T][HD]
  const __bf16* kp = kg  + bh * kT * kHD;   // [T][HD]
  const __bf16* vt = vtg + bh * kHD * kT;   // [HD][T]
  const int qw = blockIdx.x * 128 + wave * 16;

  v16bf qa0 = load_A_rowmajor(q + (size_t)qw * kHD,      kHD, lane);
  v16bf qa1 = load_A_rowmajor(q + (size_t)qw * kHD + 32, kHD, lane);

  v8f acc[4];
  acc[0] = zero8(); acc[1] = zero8(); acc[2] = zero8(); acc[3] = zero8();
  float m_i[8], l_i[8];
#pragma unroll
  for (int v = 0; v < 8; ++v) { m_i[v] = -1e30f; l_i[v] = 0.f; }

  const int n  = lane & 15;
  const int mb = (lane >> 4) << 3;
  const int nkt = (qw + 16 + 31) >> 5;  // causal tile count

  for (int j = 0; j < nkt; ++j) {
    const int kt = j * 32;

    if (j + 1 < nkt) {  // prefetch next K/V tiles (global_prefetch_b8)
      __builtin_prefetch((const void*)(kp + (size_t)(kt + 32) * kHD), 0, 1);
      __builtin_prefetch((const void*)(vt + kt + 32), 0, 1);
    }

    // S = Q K^T : K^T element (K=d, N=key) at kp[key*HD + d] -> N-major.
    v16bf b00 = load_B_nmajor(kp + (size_t)(kt +  0) * kHD +  0, kHD, lane);
    v16bf b01 = load_B_nmajor(kp + (size_t)(kt +  0) * kHD + 32, kHD, lane);
    v16bf b10 = load_B_nmajor(kp + (size_t)(kt + 16) * kHD +  0, kHD, lane);
    v16bf b11 = load_B_nmajor(kp + (size_t)(kt + 16) * kHD + 32, kHD, lane);
    v8f s0 = zero8(), s1 = zero8();
    s0 = wmma_bf16(qa0, b00, s0); s0 = wmma_bf16(qa1, b01, s0);
    s1 = wmma_bf16(qa0, b10, s1); s1 = wmma_bf16(qa1, b11, s1);

    // Online softmax over 16x32 tile; row stats per lane-half.
    float p0[8], p1[8];
#pragma unroll
    for (int v = 0; v < 8; ++v) {
      const int qrow = qw + mb + v;
      float x0 = s0[v]; if (kt + n      > qrow) x0 = -1e30f;
      float x1 = s1[v]; if (kt + 16 + n > qrow) x1 = -1e30f;
      float mt = half16_reduce_max(fmaxf(x0, x1));
      float mn = fmaxf(m_i[v], mt);
      float sc = __expf(m_i[v] - mn);
      p0[v] = __expf(x0 - mn);
      p1[v] = __expf(x1 - mn);
      float r = half16_reduce_sum(p0[v] + p1[v]);
      l_i[v] = l_i[v] * sc + r;
      m_i[v] = mn;
      acc[0][v] *= sc; acc[1][v] *= sc; acc[2][v] *= sc; acc[3][v] *= sc;
    }

    // Re-layout P (C layout -> A fragment) through per-wave LDS scratch.
    __bf16* lp = lds_p[wave];
#pragma unroll
    for (int v = 0; v < 8; ++v) {
      int m = mb + v;
      lp[m * 32 + n]      = (__bf16)p0[v];
      lp[m * 32 + 16 + n] = (__bf16)p1[v];
    }
    v16bf pa = load_A_rowmajor(lp, 32, lane);

    // O += P V : V^T stored [HD][T] -> element (K=key, N=d) at vt[d*T+key].
#pragma unroll
    for (int nc = 0; nc < 4; ++nc) {
      v16bf bv = load_B_nmajor(vt + (size_t)(nc * 16) * kT + kt, kT, lane);
      acc[nc] = wmma_bf16(pa, bv, acc[nc]);
    }
  }

  // Normalize, write y bf16 [B,T,C]; one row base per row.
#pragma unroll
  for (int v = 0; v < 8; ++v) {
    float inv = 1.f / l_i[v];
    __bf16* rp = yg + ((size_t)b * kT + (qw + mb + v)) * kC + h * kHD + n;
#pragma unroll
    for (int nc = 0; nc < 4; ++nc) rp[nc * 16] = (__bf16)(acc[nc][v] * inv);
  }
}

// ---------------------------------------------------------------------------
// Host-side launch
// ---------------------------------------------------------------------------
extern "C" void kernel_launch(void* const* d_in, const int* in_sizes, int n_in,
                              void* d_out, int out_size, void* d_ws, size_t ws_size,
                              hipStream_t stream) {
  (void)in_sizes; (void)n_in; (void)out_size; (void)ws_size;

  const float* x     = (const float*)d_in[0];
  const float* cosb  = (const float*)d_in[1];
  const float* sinb  = (const float*)d_in[2];
  const float* Wattn = (const float*)d_in[3];
  const float* Wproj = (const float*)d_in[4];
  float* out = (float*)d_out;

  char* ws = (char*)d_ws;
  size_t off = 0;
  auto alloc = [&](size_t bytes) -> void* {
    void* p = ws + off;
    off = (off + bytes + 255) & ~(size_t)255;
    return p;
  };

  const size_t nX    = (size_t)kB * kT * kC;          // 4,194,304
  const size_t nWa   = (size_t)kC * 3 * kC;           // 3,145,728
  const size_t nWp   = (size_t)kC * kC;               // 1,048,576
  const size_t nQKV  = (size_t)kB * kT * 3 * kC;      // 12,582,912
  const size_t nHead = (size_t)kB * kNH * kT * kHD;   // 4,194,304

  __bf16* x_bf  = (__bf16*)alloc(nX * 2);
  __bf16* wa_t  = (__bf16*)alloc(nWa * 2);   // [3C][C]  (N-major)
  __bf16* wp_t  = (__bf16*)alloc(nWp * 2);   // [C][C]   (N-major)
  float*  qkv   = (float*)alloc(nQKV * 4);
  __bf16* q_bf  = (__bf16*)alloc(nHead * 2);
  __bf16* k_bf  = (__bf16*)alloc(nHead * 2);
  __bf16* vt_bf = (__bf16*)alloc(nHead * 2);
  __bf16* y_bf  = (__bf16*)alloc(nX * 2);

  // 1) casts (+ one-time weight transposes to N-major)
  cast_f32_to_bf16<<<(int)(nX / 256), 256, 0, stream>>>(x, x_bf, (int)nX);
  cast_transpose_f32_bf16<<<dim3(3 * kC / 32, kC / 32), 256, 0, stream>>>(
      Wattn, wa_t, kC, 3 * kC);
  cast_transpose_f32_bf16<<<dim3(kC / 32, kC / 32), 256, 0, stream>>>(
      Wproj, wp_t, kC, kC);

  // 2) qkv = x @ W_attn   (M=4096, N=3072, K=1024)
  gemm_bf16_tn<<<dim3(3 * kC / 128, (kB * kT) / 128), 256, 0, stream>>>(
      x_bf, wa_t, qkv, kB * kT, 3 * kC, kC);

  // 3) RoPE + head split + v transpose
  const size_t nRope = (size_t)kB * kT * kNH * (kHD / 2);  // 2,097,152
  rope_split_kernel<<<(int)(nRope / 256), 256, 0, stream>>>(
      qkv, cosb, sinb, q_bf, k_bf, vt_bf);

  // 4) causal flash attention
  flash_attn_kernel<<<dim3(kT / 128, kNH, kB), 256, 0, stream>>>(
      q_bf, k_bf, vt_bf, y_bf);

  // 5) out = y @ W_proj   (M=4096, N=1024, K=1024)
  gemm_bf16_tn<<<dim3(kC / 128, (kB * kT) / 128), 256, 0, stream>>>(
      y_bf, wp_t, out, kB * kT, kC, kC);
}
